// GaussianEmbedding_38019050504715
// MI455X (gfx1250) — matrix operand order repaired and checked by reading, not compiled, not run on hardware
//
#include <hip/hip_runtime.h>

typedef __attribute__((ext_vector_type(16))) _Float16 v16h;
typedef __attribute__((ext_vector_type(8)))  _Float16 v8h;
typedef __attribute__((ext_vector_type(8)))  float    v8f;

#define NCC   128
#define NAA   65
#define KK    128
#define PD    64
#define ED    256
#define MPAD  80      // 65 rows padded to 5 tiles of 16
#define AS    136     // LDS row stride in halves (128 + 8 pad), 272B (16B-aligned)

// ---- output layout (concatenated flat, reference return order) ----
#define ATOM_OFF  0L
#define ATTN_OFF  ((long)NCC*NAA*ED)                    // 2,129,920
#define DELTA_OFF (ATTN_OFF + (long)NCC*NAA*NAA*PD)    // 36,741,120

// One-time per launch: f32 -> f16 weight staging into workspace.
__global__ void ge_convert_weights(const float* __restrict__ w1,
                                   const float* __restrict__ w2,
                                   _Float16* __restrict__ w1h,
                                   _Float16* __restrict__ w2h) {
  int idx = blockIdx.x * 256 + threadIdx.x;
  if (idx < KK * KK) w1h[idx] = (_Float16)w1[idx];
  if (idx < PD * KK) w2h[idx] = (_Float16)w2[idx];
}

__launch_bounds__(256)
__global__ void ge_main(const float* __restrict__ coord,
                        const float* __restrict__ atom_feature,
                        const int*   __restrict__ edge_type,
                        const float* __restrict__ edge_feature,
                        const float* __restrict__ mul_w,
                        const float* __restrict__ bias_w,
                        const float* __restrict__ mean,
                        const float* __restrict__ b1,
                        const float* __restrict__ b2,
                        const float* __restrict__ ep_w,
                        const float* __restrict__ ep_b,
                        const _Float16* __restrict__ w1h,
                        const _Float16* __restrict__ w2h,
                        float* __restrict__ out) {
  __shared__ __align__(16) _Float16 Ag[MPAD * AS];   // gbf tile (f16)
  __shared__ __align__(16) _Float16 Hs[MPAD * AS];   // gelu(h) tile (f16)
  __shared__ float meanL[KK];
  __shared__ float sumg[KK];                         // f32 column sums of gbf
  __shared__ float xs[NAA];                          // mul*dist + bias per neighbor

  const int t  = threadIdx.x;
  const int ci = blockIdx.x;            // c*NAA + i
  const int c  = ci / NAA;
  const int i  = ci - c * NAA;

  float* atom_out  = out + ATOM_OFF;
  float* attn_out  = out + ATTN_OFF;
  float* delta_out = out + DELTA_OFF;

  const float stdv     = mean[1] - mean[0];
  const float inv_std  = 1.0f / stdv;
  const float inv_astd = 1.0f / (sqrtf(2.0f * 3.14159f) * stdv);

  if (t < KK) { meanL[t] = mean[t]; sumg[t] = 0.0f; }

  // ---- Phase A: delta_pos, dist, per-pair gaussian params -> xs[j] ----
  if (t < NAA) {
    const float xi = coord[(c * NAA + i) * 3 + 0];
    const float yi = coord[(c * NAA + i) * 3 + 1];
    const float zi = coord[(c * NAA + i) * 3 + 2];
    const float xj = coord[(c * NAA + t) * 3 + 0];
    const float yj = coord[(c * NAA + t) * 3 + 1];
    const float zj = coord[(c * NAA + t) * 3 + 2];
    const float dx = xj - xi, dy = yj - yi, dz = zj - zi;
    const long db = ((long)ci * NAA + t) * 3;
    delta_out[db + 0] = dx;
    delta_out[db + 1] = dy;
    delta_out[db + 2] = dz;
    const float dist = sqrtf(dx * dx + dy * dy + dz * dz);
    const long eb = ((long)ci * NAA + t) * 2;
    const int t0 = edge_type[eb + 0];
    const int t1 = edge_type[eb + 1];
    const float mul = fabsf(mul_w[t0]) + fabsf(mul_w[t1]);
    const float bia = bias_w[t0] + bias_w[t1];
    xs[t] = mul * dist + bia;
  }
  __syncthreads();

  // ---- Phase B: gbf[j][k] -> LDS (f16), f32 column sums ----
  {
    const int k  = t & (KK - 1);
    const int rg = t >> 7;                 // 0 or 1: two threads per column
    const float mk = meanL[k];
    float part = 0.0f;
    for (int m = rg; m < MPAD; m += 2) {
      float g = 0.0f;
      if (m < NAA) {
        const float u = (xs[m] - mk) * inv_std;
        g = __expf(-0.5f * u * u) * inv_astd;
        part += g;
      }
      Ag[m * AS + k] = (_Float16)g;
    }
    atomicAdd(&sumg[k], part);
  }
  __syncthreads();

  const int wave = t >> 5;
  const int lane = t & 31;
  const int hf   = lane >> 4;     // half-wave select
  const int lm   = lane & 15;

  // ---- Phase C: GEMM1 H = gbf @ w1^T, exact GELU, +b1, restage f16 ----
  for (int tile = wave; tile < 40; tile += 8) {
    const int mt = tile >> 3;     // 0..4
    const int nt = tile & 7;      // 0..7
    v8f acc = {};
#pragma unroll
    for (int ks = 0; ks < 4; ks++) {
      const int K0 = ks * 32;
      union { v16h v; v8h h[2]; } au, bu;
      const int row = mt * 16 + lm;
      au.h[0] = *(const v8h*)&Ag[row * AS + K0 + hf * 8];
      au.h[1] = *(const v8h*)&Ag[row * AS + K0 + 16 + hf * 8];
      const int n = nt * 16 + lm;
      bu.h[0] = *(const v8h*)&w1h[n * KK + K0 + hf * 8];
      bu.h[1] = *(const v8h*)&w1h[n * KK + K0 + 16 + hf * 8];
      acc = __builtin_amdgcn_wmma_f32_16x16x32_f16(
          false, au.v, false, bu.v, (short)0, acc, false, false);
    }
    const int n = nt * 16 + lm;
    const float b1n = b1[n];
#pragma unroll
    for (int r = 0; r < 8; r++) {
      const int row = mt * 16 + r + 8 * hf;
      const float x = acc[r];
      const float g = 0.5f * x * (1.0f + erff(x * 0.70710678f)) + b1n;
      Hs[row * AS + n] = (_Float16)g;
    }
  }
  __syncthreads();

  // ---- Phase D: GEMM2 attn = H @ w2^T + b2 + edge_feature (fused store) ----
  const long ebase = (long)ci * NAA * PD;
  for (int tile = wave; tile < 20; tile += 8) {
    const int mt = tile >> 2;     // 0..4
    const int nt = tile & 3;      // 0..3
    v8f acc = {};
#pragma unroll
    for (int ks = 0; ks < 4; ks++) {
      const int K0 = ks * 32;
      union { v16h v; v8h h[2]; } au, bu;
      const int row = mt * 16 + lm;
      au.h[0] = *(const v8h*)&Hs[row * AS + K0 + hf * 8];
      au.h[1] = *(const v8h*)&Hs[row * AS + K0 + 16 + hf * 8];
      const int n = nt * 16 + lm;
      bu.h[0] = *(const v8h*)&w2h[n * KK + K0 + hf * 8];
      bu.h[1] = *(const v8h*)&w2h[n * KK + K0 + 16 + hf * 8];
      acc = __builtin_amdgcn_wmma_f32_16x16x32_f16(
          false, au.v, false, bu.v, (short)0, acc, false, false);
    }
    const int n = nt * 16 + lm;
    const float b2n = b2[n];
#pragma unroll
    for (int r = 0; r < 8; r++) {
      const int j = mt * 16 + r + 8 * hf;
      if (j < NAA) {
        const long o = ebase + (long)j * PD + n;
        attn_out[o] = acc[r] + b2n + edge_feature[o];
      }
    }
  }

  // ---- Phase E: atom_out = atom_feature + (sum_j gbf) @ ep_w^T + ep_b ----
  {
    const int e = t;                       // 0..255 == EMBED_DIM
    float s = ep_b[e];
#pragma unroll 8
    for (int k = 0; k < KK; k++) s += sumg[k] * ep_w[e * KK + k];
    const long o = (long)ci * ED + e;
    atom_out[o] = atom_feature[o] + s;
  }
}

extern "C" void kernel_launch(void* const* d_in, const int* in_sizes, int n_in,
                              void* d_out, int out_size, void* d_ws, size_t ws_size,
                              hipStream_t stream) {
  const float* coord        = (const float*)d_in[0];
  const float* atom_feature = (const float*)d_in[1];
  const int*   edge_type    = (const int*)  d_in[2];
  const float* edge_feature = (const float*)d_in[3];
  const float* mul_w        = (const float*)d_in[4];
  const float* bias_w       = (const float*)d_in[5];
  const float* mean         = (const float*)d_in[6];
  const float* w1           = (const float*)d_in[7];
  const float* b1           = (const float*)d_in[8];
  const float* w2           = (const float*)d_in[9];
  const float* b2           = (const float*)d_in[10];
  const float* ep_w         = (const float*)d_in[11];
  const float* ep_b         = (const float*)d_in[12];

  _Float16* w1h = (_Float16*)d_ws;            // 128*128 halves = 32 KB
  _Float16* w2h = w1h + KK * KK;              // 64*128 halves  = 16 KB

  ge_convert_weights<<<(KK * KK + 255) / 256, 256, 0, stream>>>(w1, w2, w1h, w2h);

  float* out = (float*)d_out;
  ge_main<<<NCC * NAA, 256, 0, stream>>>(coord, atom_feature, edge_type,
                                         edge_feature, mul_w, bias_w, mean,
                                         b1, b2, ep_w, ep_b, w1h, w2h, out);
}
